// RvT_91190745629008
// MI455X (gfx1250) — matrix operand order repaired and checked by reading, not compiled, not run on hardware
//
#include <hip/hip_runtime.h>
#include <math.h>

typedef __attribute__((ext_vector_type(16))) __bf16 v16bf;
typedef __attribute__((ext_vector_type(8)))  __bf16 v8bf_t;
typedef __attribute__((ext_vector_type(8)))  float  v8f;
typedef __attribute__((ext_vector_type(4)))  unsigned int v4u;
typedef __attribute__((ext_vector_type(8)))  int v8i;
typedef __attribute__((ext_vector_type(4)))  int v4i;

#define DEVINL __device__ __forceinline__
#define USE_CDNA5_ASYNC 1

static constexpr int Nn = 8, Cin = 128, HW = 1296, Cc = 256, Gg = 8;
static constexpr int Dd = 1156;            // 34*34 key positions
static constexpr int DTILES = 74;          // DPAD/16
static constexpr int DPAD = 1184;          // 74*16 = 37*32

DEVINL unsigned short f2bf(float f) {
    union { float f; unsigned u; } v; v.f = f;
    unsigned r = v.u + 0x7fffu + ((v.u >> 16) & 1u);
    return (unsigned short)(r >> 16);
}

DEVINL v8f wmma_bf16(v16bf a, v16bf b, v8f c) {
    return __builtin_amdgcn_wmma_f32_16x16x32_bf16(
        false, a, false, b, (short)0, c, false, false);
}

// Load a 16-element fragment as two aligned 16B vector loads.
DEVINL v16bf ldfrag(const unsigned short* p0, const unsigned short* p1) {
    v8bf_t a = *(const v8bf_t*)p0;
    v8bf_t b = *(const v8bf_t*)p1;
    return __builtin_shufflevector(a, b, 0, 1, 2, 3, 4, 5, 6, 7,
                                   8, 9, 10, 11, 12, 13, 14, 15);
}

#if USE_CDNA5_ASYNC
// LDS byte offset of a __shared__ object (generic -> AS3 -> int)
#define LDS_OFF(p) ((unsigned)(unsigned long long)(__attribute__((address_space(3))) const void*)(p))
#endif

// ---------------- weight conversion / packing ----------------
__global__ void k_cvt(const float* __restrict__ in, unsigned short* __restrict__ out, int n) {
    int i = blockIdx.x * 256 + threadIdx.x;
    if (i < n) out[i] = f2bf(in[i]);
}

__global__ void k_gatepack(const float* Wi1, const float* Wi2,
                           const float* Wf1, const float* Wf2,
                           const float* Wg1, const float* Wg2,
                           const float* Wo1, const float* Wo2,
                           unsigned short* __restrict__ gw) {
    int i = blockIdx.x * 256 + threadIdx.x;
    const int total = Gg * 4 * 32 * 608;
    if (i >= total) return;
    int col = i % 608; int t = i / 608;
    int oc = t % 32; t /= 32;
    int gate = t % 4; int g = t / 4;
    const float* W1 = (gate == 0) ? Wi1 : (gate == 1) ? Wf1 : (gate == 2) ? Wg1 : Wo1;
    const float* W2 = (gate == 0) ? Wi2 : (gate == 1) ? Wf2 : (gate == 2) ? Wg2 : Wo2;
    int c = g * 32 + oc;
    float v = (col < 32) ? W1[c * 32 + col] : W2[c * 576 + (col - 32)];
    gw[i] = f2bf(v);
}

__global__ void k_xh_h0(const float* __restrict__ h0, unsigned short* __restrict__ xh) {
    int i = blockIdx.x * 256 + threadIdx.x;
    const int total = Nn * Cc * HW;
    if (i >= total) return;
    int p = i % HW; int t = i / HW;
    int c = t % Cc; int n = t / Cc;
    int ch = (c >> 5) * 64 + 32 + (c & 31);
    xh[(n * 512 + ch) * HW + p] = f2bf(h0[i]);
}

__global__ void k_padzero(unsigned short* __restrict__ kT, unsigned short* __restrict__ vp) {
    int i = blockIdx.x * 256 + threadIdx.x;
    const int nk = 64 * 28 * 32;
    if (i < nk) {
        int c = i & 31; int t = i >> 5;
        int d = Dd + (t % 28); int ng = t / 28;
        kT[((size_t)ng * DPAD + d) * 32 + c] = 0;
    } else if (i < 2 * nk) {
        int j = i - nk;
        int dc = j % 28; int t = j / 28;
        int c = t & 31; int ng = t >> 5;
        vp[((size_t)(ng * 32 + c)) * DPAD + Dd + dc] = 0;
    }
}

// ---------------- 1x1 input projection as WMMA GEMM ----------------
static constexpr int XSTR = 136;
__global__ void __launch_bounds__(256) k_xproj(const float* __restrict__ x,
                                               const unsigned short* __restrict__ wx,
                                               unsigned short* __restrict__ xh) {
    __shared__ unsigned short As[16 * XSTR];
    __shared__ unsigned short Bt[128 * XSTR];
    int tid = threadIdx.x, lane = tid & 31, wv = tid >> 5;
    int n = blockIdx.z, mt = blockIdx.y, bx = blockIdx.x;
    for (int i = tid; i < 16 * 128; i += 256) {
        int m = i >> 7, k = i & 127;
        As[m * XSTR + k] = wx[(mt * 16 + m) * 128 + k];
    }
    int p0 = bx * 128;
    for (int i = tid; i < 64 * 128; i += 256) {
        int pl = i & 127, k0 = (i >> 7) * 2;
        int p = p0 + pl;
        unsigned short e0 = 0, e1 = 0;
        if (p < HW) {
            e0 = f2bf(x[(n * Cin + k0) * HW + p]);
            e1 = f2bf(x[(n * Cin + k0 + 1) * HW + p]);
        }
        *(unsigned*)&Bt[pl * XSTR + k0] = (unsigned)e0 | ((unsigned)e1 << 16);
    }
    __syncthreads();
    int nt = bx * 8 + wv;
    if (nt >= 81) return;
    int half = lane >> 4, l16 = lane & 15;
    const unsigned short* arow = As + l16 * XSTR;
    const unsigned short* brow = Bt + (wv * 16 + l16) * XSTR;
    v8f acc = {};
    for (int kc = 0; kc < 4; kc++) {
        v16bf af = ldfrag(arow + kc * 32 + 8 * half, arow + kc * 32 + 16 + 8 * half);
        v16bf bm = ldfrag(brow + kc * 32 + 8 * half, brow + kc * 32 + 16 + 8 * half);
        acc = wmma_bf16(af, bm, acc);
    }
    int p = nt * 16 + l16;
#pragma unroll
    for (int r = 0; r < 8; r++) {
        int c = mt * 16 + r + 8 * half;
        int ch = (c >> 5) * 64 + (c & 31);
        xh[(n * 512 + ch) * HW + p] = f2bf(acc[r]);
    }
}

// ---------------- grouped 3x3 conv (implicit GEMM, K=576) ----------------
static constexpr int BSTR = 584;
template <bool SAME>
__global__ void __launch_bounds__(256) k_gconv3(const unsigned short* __restrict__ xh,
                                                const unsigned short* __restrict__ wgt,
                                                unsigned short* __restrict__ outN,
                                                unsigned short* __restrict__ outT,
                                                int OH, int OW, int OS) {
    __shared__ unsigned short patch[64 * 100];
    __shared__ unsigned short Bp[64 * BSTR];
    int tid = threadIdx.x, lane = tid & 31, wv = tid >> 5;
    int tx = blockIdx.x, ty = blockIdx.y, ng = blockIdx.z;
    int n = ng >> 3, g = ng & 7;
    int iy0 = ty * 8 - (SAME ? 1 : 0), ix0 = tx * 8 - (SAME ? 1 : 0);

#if USE_CDNA5_ASYNC
    if (!SAME) {
        // VALID path: one TDM descriptor moves the whole 10x10x64ch bf16 tile
        // into LDS; x/y overhang past the 36x36 tensor is zero-filled by TDM OOB.
        if (wv == 0) {
            const unsigned short* gsrc = xh + (size_t)(n * 512 + g * 64) * HW + iy0 * 36 + ix0;
            unsigned long long ga = (unsigned long long)gsrc;
            unsigned lds = LDS_OFF(patch);
            v4u g0;
            g0[0] = 1u;                                            // count=1, user mode
            g0[1] = lds;                                           // lds_addr
            g0[2] = (unsigned)(ga & 0xffffffffu);                  // global_addr[31:0]
            g0[3] = (unsigned)((ga >> 32) & 0x01ffffffu) | (2u << 30); // addr[56:32], type=2
            v8i g1;
            g1[0] = (1 << 16);                                     // data_size=2B, no multicast
            g1[1] = (int)(36u << 16);                              // tensor_dim0 lo16 = 36
            g1[2] = (int)(36u << 16);                              // dim0 hi16=0 | dim1 lo16=36
            g1[3] = (int)(10u << 16);                              // dim1 hi16=0 | tile_dim0=10
            g1[4] = (int)(10u | (64u << 16));                      // tile_dim1=10 | tile_dim2=64
            g1[5] = 36;                                            // tensor_dim0_stride lo32
            g1[6] = (int)(1296u << 16);                            // d0s hi16=0 | d1_stride lo16
            g1[7] = 0;                                             // d1_stride hi32 (1296>>16)
            v4i g2; g2[0] = 64; g2[1] = 0; g2[2] = 0; g2[3] = 0;   // tensor_dim2=64
            v4i g3; g3[0] = 0; g3[1] = 0; g3[2] = 0; g3[3] = 0;
            asm volatile("tensor_load_to_lds %0, %1, %2, %3"
                         :: "s"(g0), "s"(g1), "s"(g2), "s"(g3) : "memory");
            asm volatile("s_wait_tensorcnt 0x0" ::: "memory");
        }
    } else
#endif
    {
        for (int i = tid; i < 6400; i += 256) {
            int ic = i / 100, pos = i % 100, py = pos / 10, px = pos % 10;
            int iy = iy0 + py, ix = ix0 + px;
            unsigned short v = 0;
            if (iy >= 0 && iy < 36 && ix >= 0 && ix < 36)
                v = xh[(n * 512 + g * 64 + ic) * HW + iy * 36 + ix];
            patch[i] = v;
        }
    }
    __syncthreads();
    for (int i = tid; i < 64 * 288; i += 256) {
        int p = i / 288, k0 = (i % 288) * 2;
        int oy_l = p >> 3, ox_l = p & 7;
        int k = k0;     int ic = k / 9, kk = k % 9;
        unsigned short e0 = patch[ic * 100 + (oy_l + kk / 3) * 10 + (ox_l + kk % 3)];
        k = k0 + 1;     ic = k / 9; kk = k % 9;
        unsigned short e1 = patch[ic * 100 + (oy_l + kk / 3) * 10 + (ox_l + kk % 3)];
        *(unsigned*)&Bp[p * BSTR + k0] = (unsigned)e0 | ((unsigned)e1 << 16);
    }
    __syncthreads();
    int mt = wv & 1, s = wv >> 1;
    int half = lane >> 4, l16 = lane & 15;
    const unsigned short* arow = wgt + (g * 32 + mt * 16 + l16) * 576;
    const unsigned short* brow = Bp + (s * 16 + l16) * BSTR;
    v8f acc = {};
    for (int kc = 0; kc < 18; kc++) {
        __builtin_prefetch(arow + kc * 32 + 64, 0, 1);
        v16bf af = ldfrag(arow + kc * 32 + 8 * half, arow + kc * 32 + 16 + 8 * half);
        v16bf bm = ldfrag(brow + kc * 32 + 8 * half, brow + kc * 32 + 16 + 8 * half);
        acc = wmma_bf16(af, bm, acc);
    }
    int pl = s * 16 + l16;
    int oy = ty * 8 + (pl >> 3), ox = tx * 8 + (pl & 7);
    if (oy < OH && ox < OW) {
        int p = oy * OW + ox;
#pragma unroll
        for (int r = 0; r < 8; r++) {
            int oc = mt * 16 + r + 8 * half;
            unsigned short h = f2bf(acc[r]);
            if (outN) outN[(size_t)(ng * 32 + oc) * OS + p] = h;
            if (outT) outT[((size_t)ng * DPAD + p) * 32 + oc] = h;
        }
    }
}

// ---------------- attention ----------------
static constexpr int PSTR = 1192;
static constexpr int QSTR = 40;
__global__ void __launch_bounds__(256) k_attn(const unsigned short* __restrict__ qb,
                                              const unsigned short* __restrict__ kT,
                                              const unsigned short* __restrict__ vp,
                                              const float* __restrict__ tau,
                                              unsigned short* __restrict__ ab) {
    __shared__ float lg[16 * DPAD];
    __shared__ unsigned short pbt[16 * PSTR];
    __shared__ unsigned short qs[16 * QSTR];
    __shared__ float red[16 * 16];
    __shared__ float rowv[16];
    __shared__ float part[8 * 256];
    int tid = threadIdx.x, lane = tid & 31, wv = tid >> 5;
    int qt = blockIdx.x, ng = blockIdx.y;
    int n = ng >> 3, g = ng & 7;
    float tv = tau[g];
    int half = lane >> 4, l16 = lane & 15;
    const unsigned short* qg = qb + (size_t)ng * 32 * HW;

    for (int i = tid; i < 512; i += 256) {
        int qp = i & 15, c = i >> 4;
        qs[qp * QSTR + c] = qg[c * HW + qt * 16 + qp];
    }
    __syncthreads();

    const unsigned short* qrow = qs + l16 * QSTR;
    v16bf af1 = ldfrag(qrow + 8 * half, qrow + 16 + 8 * half);
    for (int dt = wv; dt < DTILES; dt += 8) {
        const unsigned short* krow = kT + ((size_t)ng * DPAD + dt * 16 + l16) * 32;
        v16bf bm = ldfrag(krow + 8 * half, krow + 16 + 8 * half);
        v8f t = {};
        t = wmma_bf16(af1, bm, t);
        int d = dt * 16 + l16;
#pragma unroll
        for (int r = 0; r < 8; r++)
            lg[(r + 8 * half) * DPAD + d] = (d < Dd) ? tv * t[r] : -1e30f;
    }
    __syncthreads();

    int row = tid >> 4, c16 = tid & 15;
    float mx = -1e30f;
    for (int j = c16; j < DPAD; j += 16) mx = fmaxf(mx, lg[row * DPAD + j]);
    red[row * 16 + c16] = mx;
    __syncthreads();
    if (c16 == 0) {
        float m2 = -1e30f;
        for (int j = 0; j < 16; j++) m2 = fmaxf(m2, red[row * 16 + j]);
        rowv[row] = m2;
    }
    __syncthreads();
    float M = rowv[row];
    float sm = 0.f;
    for (int j = c16; j < DPAD; j += 16) sm += __expf(lg[row * DPAD + j] - M);
    __syncthreads();
    red[row * 16 + c16] = sm;
    __syncthreads();
    if (c16 == 0) {
        float s2 = 0.f;
        for (int j = 0; j < 16; j++) s2 += red[row * 16 + j];
        rowv[row] = s2;
    }
    __syncthreads();
    float inv = 1.f / rowv[row];
    for (int j = c16; j < DPAD; j += 16)
        pbt[row * PSTR + j] = f2bf(__expf(lg[row * DPAD + j] - M) * inv);
    __syncthreads();

    int ct = wv & 1, kp = wv >> 1;
    const unsigned short* vrow = vp + ((size_t)(ng * 32 + ct * 16 + l16)) * DPAD;
    const unsigned short* prow = pbt + l16 * PSTR;
    v8f acc = {};
    for (int kc = kp; kc < DPAD / 32; kc += 4) {
        v16bf af = ldfrag(vrow + kc * 32 + 8 * half, vrow + kc * 32 + 16 + 8 * half);
        v16bf bm = ldfrag(prow + kc * 32 + 8 * half, prow + kc * 32 + 16 + 8 * half);
        acc = wmma_bf16(af, bm, acc);
    }
#pragma unroll
    for (int r = 0; r < 8; r++) part[wv * 256 + r * 32 + lane] = acc[r];
    __syncthreads();
    for (int f = tid; f < 512; f += 256) {
        int ct2 = f >> 8, j = f & 255;
        float s = part[(0 * 2 + ct2) * 256 + j] + part[(1 * 2 + ct2) * 256 + j] +
                  part[(2 * 2 + ct2) * 256 + j] + part[(3 * 2 + ct2) * 256 + j];
        int r = j >> 5, L = j & 31;
        int m = r + 8 * (L >> 4);
        int qp = qt * 16 + (L & 15);
        ab[((size_t)n * 256 + g * 32 + ct2 * 16 + m) * HW + qp] = f2bf(s);
    }
}

// ---------------- fused gates (4 GEMMs, K=608) + LSTM pointwise ----------------
static constexpr int GSTR = 616;
__global__ void __launch_bounds__(256) k_gates(const unsigned short* __restrict__ xh,
                                               const unsigned short* __restrict__ ab,
                                               const unsigned short* __restrict__ gw,
                                               const float* __restrict__ bi,
                                               const float* __restrict__ bfv,
                                               const float* __restrict__ bg,
                                               const float* __restrict__ bo,
                                               const float* __restrict__ c0,
                                               float* __restrict__ hout) {
    __shared__ unsigned short patch[64 * 100];
    __shared__ unsigned short as_[32 * 64];
    __shared__ unsigned short Bp[64 * GSTR];
    __shared__ float zb[128 * 64];
    int tid = threadIdx.x, lane = tid & 31, wv = tid >> 5;
    int tx = blockIdx.x, ty = blockIdx.y, ng = blockIdx.z;
    int n = ng >> 3, g = ng & 7;
    int iy0 = ty * 8 - 1, ix0 = tx * 8 - 1;
    for (int i = tid; i < 6400; i += 256) {
        int ic = i / 100, pos = i % 100, py = pos / 10, px = pos % 10;
        int iy = iy0 + py, ix = ix0 + px;
        unsigned short v = 0;
        if (iy >= 0 && iy < 36 && ix >= 0 && ix < 36)
            v = xh[(n * 512 + g * 64 + ic) * HW + iy * 36 + ix];
        patch[i] = v;
    }
#if USE_CDNA5_ASYNC
    {
        // stage 'a' tile rows via async global->LDS b64 (4 bf16 per op)
        unsigned lbase = LDS_OFF(as_);
        for (int i = tid; i < 512; i += 256) {
            int ic = i >> 4, r8 = i & 15;
            int py = r8 >> 1, c4 = (r8 & 1) * 4;
            int oy = ty * 8 + py, ox0 = tx * 8 + c4;
            unsigned loff = lbase + (unsigned)(ic * 64 + py * 8 + c4) * 2u;
            if (oy < 36) {
                const unsigned short* src =
                    ab + ((size_t)n * 256 + g * 32 + ic) * HW + oy * 36 + ox0;
                unsigned long long ga = (unsigned long long)src;
                asm volatile("global_load_async_to_lds_b64 %0, %1, off"
                             :: "v"(loff), "v"(ga) : "memory");
            } else {
                *(unsigned long long*)&as_[ic * 64 + py * 8 + c4] = 0ull;
            }
        }
        asm volatile("s_wait_asynccnt 0x0" ::: "memory");
    }
#else
    for (int i = tid; i < 32 * 64; i += 256) {
        int ic = i >> 6, pl = i & 63;
        int oy = ty * 8 + (pl >> 3), ox = tx * 8 + (pl & 7);
        unsigned short v = 0;
        if (oy < 36 && ox < 36) v = ab[((size_t)n * 256 + g * 32 + ic) * HW + oy * 36 + ox];
        as_[i] = v;
    }
#endif
    __syncthreads();
    for (int i = tid; i < 64 * 304; i += 256) {
        int p = i / 304, k0 = (i % 304) * 2;
        int oy_l = p >> 3, ox_l = p & 7;
        unsigned short e0, e1;
        if (k0 < 32) {
            e0 = as_[k0 * 64 + p];
            e1 = as_[(k0 + 1) * 64 + p];
        } else {
            int k = k0 - 32; int ic = k / 9, kk = k % 9;
            e0 = patch[ic * 100 + (oy_l + kk / 3) * 10 + (ox_l + kk % 3)];
            k = k0 - 31; ic = k / 9; kk = k % 9;
            e1 = patch[ic * 100 + (oy_l + kk / 3) * 10 + (ox_l + kk % 3)];
        }
        *(unsigned*)&Bp[p * GSTR + k0] = (unsigned)e0 | ((unsigned)e1 << 16);
    }
    __syncthreads();
    int gate = wv >> 1, halfM = wv & 1;
    int half = lane >> 4, l16 = lane & 15;
    const unsigned short* arow = gw + ((g * 4 + gate) * 32 + halfM * 16 + l16) * 608;
    for (int s = 0; s < 4; s++) {
        const unsigned short* brow = Bp + (s * 16 + l16) * GSTR;
        v8f acc = {};
        for (int kc = 0; kc < 19; kc++) {
            __builtin_prefetch(arow + kc * 32 + 64, 0, 1);
            v16bf af = ldfrag(arow + kc * 32 + 8 * half, arow + kc * 32 + 16 + 8 * half);
            v16bf bm = ldfrag(brow + kc * 32 + 8 * half, brow + kc * 32 + 16 + 8 * half);
            acc = wmma_bf16(af, bm, acc);
        }
        int pl = s * 16 + l16;
#pragma unroll
        for (int r = 0; r < 8; r++) {
            int oc = halfM * 16 + r + 8 * half;
            zb[(gate * 32 + oc) * 64 + pl] = acc[r];
        }
    }
    __syncthreads();
    for (int i = tid; i < 32 * 64; i += 256) {
        int oc = i >> 6, pl = i & 63;
        int oy = ty * 8 + (pl >> 3), ox = tx * 8 + (pl & 7);
        if (oy < 36 && ox < 36) {
            int cch = g * 32 + oc;
            float zi = zb[(0 * 32 + oc) * 64 + pl] + bi[cch];
            float zf = zb[(1 * 32 + oc) * 64 + pl] + bfv[cch];
            float zg = zb[(2 * 32 + oc) * 64 + pl] + bg[cch];
            float zo = zb[(3 * 32 + oc) * 64 + pl] + bo[cch];
            float ii = 1.f / (1.f + __expf(-zi));
            float ff = 1.f / (1.f + __expf(-zf));
            float gg = tanhf(zg);
            float oo = 1.f / (1.f + __expf(-zo));
            float c0v = c0[((size_t)n * 256 + cch) * HW + oy * 36 + ox];
            float cn = ff * c0v + ii * gg;
            hout[((size_t)n * 256 + cch) * HW + oy * 36 + ox] = oo * tanhf(cn);
        }
    }
}

extern "C" void kernel_launch(void* const* d_in, const int* in_sizes, int n_in,
                              void* d_out, int out_size, void* d_ws, size_t ws_size,
                              hipStream_t stream) {
    const float* x   = (const float*)d_in[0];
    const float* h0  = (const float*)d_in[1];
    const float* c0  = (const float*)d_in[2];
    const float* Wx  = (const float*)d_in[3];
    const float* Wq  = (const float*)d_in[4];
    const float* Wk  = (const float*)d_in[5];
    const float* Wv  = (const float*)d_in[6];
    const float* Wi1 = (const float*)d_in[7];
    const float* Wi2 = (const float*)d_in[8];
    const float* bi  = (const float*)d_in[9];
    const float* Wf1 = (const float*)d_in[10];
    const float* Wf2 = (const float*)d_in[11];
    const float* bf  = (const float*)d_in[12];
    const float* Wg1 = (const float*)d_in[13];
    const float* Wg2 = (const float*)d_in[14];
    const float* bg  = (const float*)d_in[15];
    const float* Wo1 = (const float*)d_in[16];
    const float* Wo2 = (const float*)d_in[17];
    const float* bo  = (const float*)d_in[18];
    const float* tau = (const float*)d_in[19];
    float* hout = (float*)d_out;

    char* base = (char*)d_ws;
    size_t off = 0;
    auto take = [&](size_t bytes) { char* p = base + off; off = (off + bytes + 255) & ~(size_t)255; return p; };
    unsigned short* xh  = (unsigned short*)take((size_t)Nn * 512 * HW * 2);
    unsigned short* qb  = (unsigned short*)take((size_t)Nn * 256 * HW * 2);
    unsigned short* kT  = (unsigned short*)take((size_t)64 * DPAD * 32 * 2);
    unsigned short* vp  = (unsigned short*)take((size_t)64 * 32 * DPAD * 2);
    unsigned short* abf = (unsigned short*)take((size_t)Nn * 256 * HW * 2);
    unsigned short* wxb = (unsigned short*)take((size_t)256 * 128 * 2);
    unsigned short* wqb = (unsigned short*)take((size_t)256 * 576 * 2);
    unsigned short* wkb = (unsigned short*)take((size_t)256 * 576 * 2);
    unsigned short* wvb = (unsigned short*)take((size_t)256 * 576 * 2);
    unsigned short* gwb = (unsigned short*)take((size_t)Gg * 4 * 32 * 608 * 2);
    (void)ws_size; (void)in_sizes; (void)n_in; (void)out_size;

    auto blocks = [](int n) { return (n + 255) / 256; };
    k_cvt<<<blocks(256 * 128), 256, 0, stream>>>(Wx, wxb, 256 * 128);
    k_cvt<<<blocks(256 * 576), 256, 0, stream>>>(Wq, wqb, 256 * 576);
    k_cvt<<<blocks(256 * 576), 256, 0, stream>>>(Wk, wkb, 256 * 576);
    k_cvt<<<blocks(256 * 576), 256, 0, stream>>>(Wv, wvb, 256 * 576);
    k_gatepack<<<blocks(Gg * 4 * 32 * 608), 256, 0, stream>>>(Wi1, Wi2, Wf1, Wf2, Wg1, Wg2, Wo1, Wo2, gwb);
    k_padzero<<<blocks(2 * 64 * 28 * 32), 256, 0, stream>>>(kT, vp);

    k_xh_h0<<<blocks(Nn * Cc * HW), 256, 0, stream>>>(h0, xh);
    k_xproj<<<dim3(11, 16, Nn), 256, 0, stream>>>(x, wxb, xh);

    k_gconv3<true ><<<dim3(5, 5, Nn * Gg), 256, 0, stream>>>(xh, wqb, qb, nullptr, 36, 36, HW);
    k_gconv3<false><<<dim3(5, 5, Nn * Gg), 256, 0, stream>>>(xh, wkb, nullptr, kT, 34, 34, 0);
    k_gconv3<false><<<dim3(5, 5, Nn * Gg), 256, 0, stream>>>(xh, wvb, vp, nullptr, 34, 34, DPAD);

    k_attn<<<dim3(81, Nn * Gg), 256, 0, stream>>>(qb, kT, vp, tau, abf);

    k_gates<<<dim3(5, 5, Nn * Gg), 256, 0, stream>>>(xh, abf, gwb, bi, bf, bg, bo, c0, hout);
}
